// SAGENet_46342697124057
// MI455X (gfx1250) — compile-verified, matrix operand order checked by reading, not run on hardware
//
#include <hip/hip_runtime.h>
#include <hip/hip_bf16.h>

#define D 128

typedef __attribute__((ext_vector_type(16))) __bf16 v16bf;
typedef __attribute__((ext_vector_type(8)))  float  v8f;

__device__ __forceinline__ void atomAddF(float* p, float v) {
    unsafeAtomicAdd(p, v);   // native global_atomic_add_f32 on gfx1250
}

// ---------------- utility kernels ----------------

__global__ void zero_kernel(float* p, long n) {
    long i = (long)blockIdx.x * blockDim.x + threadIdx.x;
    long stride = (long)gridDim.x * blockDim.x;
    for (; i < n; i += stride) p[i] = 0.0f;
}

__global__ void degree_kernel(const int* __restrict__ dst, float* __restrict__ deg, int E) {
    int e = blockIdx.x * blockDim.x + threadIdx.x;
    if (e < E) atomAddF(&deg[dst[e]], 1.0f);
}

__global__ void invdeg_kernel(const float* __restrict__ deg, float* __restrict__ invdeg, int n) {
    int i = blockIdx.x * blockDim.x + threadIdx.x;
    if (i < n) invdeg[i] = 1.0f / fmaxf(deg[i], 1.0f);
}

__global__ void vecadd_kernel(const float* __restrict__ a, const float* __restrict__ b,
                              float* __restrict__ c, int n) {
    int i = blockIdx.x * blockDim.x + threadIdx.x;
    if (i < n) c[i] = a[i] + b[i];
}

// one wave per edge; lane handles 4 contiguous floats -> coalesced float4 gather,
// 4 native f32 atomics per lane into the (L2-resident) accumulator
__global__ void scatter_kernel(const float* __restrict__ X, const int* __restrict__ src,
                               const int* __restrict__ dst, float* __restrict__ agg, int E) {
    int gid = blockIdx.x * blockDim.x + threadIdx.x;
    int lane = gid & 31;
    int e = gid >> 5;
    if (e >= E) return;
    long s = src[e], d = dst[e];
    const float4 v = *(const float4*)(X + s * D + lane * 4);
    float* p = agg + d * D + lane * 4;
    atomAddF(p + 0, v.x);
    atomAddF(p + 1, v.y);
    atomAddF(p + 2, v.z);
    atomAddF(p + 3, v.w);
}

// Pack a 128x128 f32 weight (optionally the sum of two) into the per-lane
// v_wmma_f32_16x16x32_bf16 B-fragment layout:
// entry index = ((ct*4 + ks)*32 + lane), 16 bf16 per entry.
// lane n' = ct*16 + (lane&15); half = lane>>4
// elem e -> K = ks*32 + (e<8 ? 0 : 16) + 8*half + (e&7)
__global__ void pack_weights_kernel(const float* __restrict__ W, const float* __restrict__ W2,
                                    __bf16* __restrict__ P) {
    int idx = blockIdx.x * blockDim.x + threadIdx.x;   // 0..1023
    if (idx >= 1024) return;
    int lane = idx & 31;
    int ks   = (idx >> 5) & 3;
    int ct   = idx >> 7;
    int n    = ct * 16 + (lane & 15);
    int half = lane >> 4;
    __bf16* p = P + (long)idx * 16;
#pragma unroll
    for (int e = 0; e < 16; ++e) {
        int k = ks * 32 + ((e < 8) ? 0 : 16) + 8 * half + (e & 7);
        float v = W[k * D + n];
        if (W2) v += W2[k * D + n];
        p[e] = (__bf16)v;
    }
}

// ---------------- fused dual-GEMM:  Out = (A1 * invdeg) @ W1 + A2 @ W2 + bias [, ELU] ---------
// Both packed weight matrices (2 x 32 KB) are staged into LDS once per block
// (cooperative copy + barrier), so WMMA B-fragments come from ds_load_b128
// instead of per-wave redundant global traffic.
// One wave computes a 16-row x 128-col output stripe: 8 v8f accumulators,
// 4 K-steps per input matrix, 64 v_wmma_f32_16x16x32_bf16 per wave.
__global__ void __launch_bounds__(256)
gemm_dual_kernel(const float* __restrict__ A1, const float* __restrict__ invdeg,
                 const float* __restrict__ A2,
                 const __bf16* __restrict__ P1, const __bf16* __restrict__ P2,
                 const float* __restrict__ bias,
                 float* __restrict__ Out, int nRowTiles, int applyElu) {
    __shared__ __bf16 ldsW[2 * D * D];          // 64 KB: [P1 | P2] in fragment order

    // cooperative 64 KB fill: 4096 uint4, 16 per thread, fully coalesced
    {
        const uint4* g1 = (const uint4*)P1;
        const uint4* g2 = (const uint4*)P2;
        uint4* l = (uint4*)ldsW;
#pragma unroll
        for (int i = 0; i < 8; ++i)
            l[threadIdx.x + i * 256] = g1[threadIdx.x + i * 256];
#pragma unroll
        for (int i = 0; i < 8; ++i)
            l[2048 + threadIdx.x + i * 256] = g2[threadIdx.x + i * 256];
    }
    __syncthreads();

    int wave = threadIdx.x >> 5;
    int lane = threadIdx.x & 31;
    int rowTile = blockIdx.x * 8 + wave;
    if (rowTile >= nRowTiles) return;              // after barrier; uniform per wave
    int half = lane >> 4;
    long m = (long)rowTile * 16 + (lane & 15);

    v8f acc[8] = {};

    const float sDeg = invdeg[m];

    for (int pass = 0; pass < 2; ++pass) {
        const float*  A = pass ? A2 : A1;
        const __bf16* Wl = ldsW + (pass ? D * D : 0);
        const float   s = pass ? 1.0f : sDeg;
        const float* arow = A + m * D;
#pragma unroll
        for (int ks = 0; ks < 4; ++ks) {
            int k0 = ks * 32 + 8 * half;           // 32B-aligned
            float4 f0 = *(const float4*)(arow + k0);
            float4 f1 = *(const float4*)(arow + k0 + 4);
            float4 f2 = *(const float4*)(arow + k0 + 16);
            float4 f3 = *(const float4*)(arow + k0 + 20);
            v16bf a;
            a[0]  = (__bf16)(f0.x * s); a[1]  = (__bf16)(f0.y * s);
            a[2]  = (__bf16)(f0.z * s); a[3]  = (__bf16)(f0.w * s);
            a[4]  = (__bf16)(f1.x * s); a[5]  = (__bf16)(f1.y * s);
            a[6]  = (__bf16)(f1.z * s); a[7]  = (__bf16)(f1.w * s);
            a[8]  = (__bf16)(f2.x * s); a[9]  = (__bf16)(f2.y * s);
            a[10] = (__bf16)(f2.z * s); a[11] = (__bf16)(f2.w * s);
            a[12] = (__bf16)(f3.x * s); a[13] = (__bf16)(f3.y * s);
            a[14] = (__bf16)(f3.z * s); a[15] = (__bf16)(f3.w * s);
#pragma unroll
            for (int ct = 0; ct < 8; ++ct) {
                v16bf b = *(const v16bf*)(Wl + ((ct * 4 + ks) * 32 + lane) * 16);
                acc[ct] = __builtin_amdgcn_wmma_f32_16x16x32_bf16(
                    false, a, false, b, (short)0, acc[ct], false, false);
            }
        }
    }

    // epilogue: C/D layout -> lane holds rows m0+8*half+r (r=0..7), col ct*16+(lane&15)
    int nb = lane & 15;
    long mbase = (long)rowTile * 16 + 8 * half;
#pragma unroll
    for (int ct = 0; ct < 8; ++ct) {
        int n = ct * 16 + nb;
        float bv = bias[n];
#pragma unroll
        for (int r = 0; r < 8; ++r) {
            float v = acc[ct][r] + bv;
            if (applyElu) v = (v > 0.0f) ? v : (__expf(v) - 1.0f);
            Out[(mbase + r) * D + n] = v;
        }
    }
}

// ---------------- BatchNorm helpers ----------------

__global__ void bn_stats_kernel(const float* __restrict__ H, float* __restrict__ sums,
                                float* __restrict__ sumsq, int n) {
    int c = threadIdx.x;                               // 128 threads/block
    int rowsPer = (n + gridDim.x - 1) / gridDim.x;
    int r0 = blockIdx.x * rowsPer;
    int r1 = min(n, r0 + rowsPer);
    float s = 0.0f, q = 0.0f;
    for (int r = r0; r < r1; ++r) {
        float v = H[(long)r * D + c];
        s += v; q += v * v;
    }
    atomAddF(&sums[c], s);
    atomAddF(&sumsq[c], q);
}

__global__ void bn_elu_kernel(const float* __restrict__ H, const float* __restrict__ sums,
                              const float* __restrict__ sumsq, const float* __restrict__ gamma,
                              const float* __restrict__ beta, float* __restrict__ Outp, int n) {
    long total = (long)n * D;
    long i = (long)blockIdx.x * blockDim.x + threadIdx.x;
    long stride = (long)gridDim.x * blockDim.x;
    float invn = 1.0f / (float)n;
    for (; i < total; i += stride) {
        int c = (int)(i & (D - 1));
        float mu  = sums[c] * invn;
        float var = sumsq[c] * invn - mu * mu;
        float v = gamma[c] * (H[i] - mu) * rsqrtf(var + 1e-5f) + beta[c];
        Outp[i] = (v > 0.0f) ? v : (__expf(v) - 1.0f);
    }
}

// ---------------- final [N,128] @ [128,1] + b ----------------
__global__ void out_kernel(const float* __restrict__ H, const float* __restrict__ W,
                           const float* __restrict__ b, float* __restrict__ out, int n) {
    int gid = blockIdx.x * blockDim.x + threadIdx.x;
    int lane = gid & 31;
    int node = gid >> 5;
    if (node >= n) return;
    float4 h4 = *(const float4*)(H + (long)node * D + lane * 4);
    float4 w4 = *(const float4*)(W + lane * 4);
    float d = h4.x * w4.x + h4.y * w4.y + h4.z * w4.z + h4.w * w4.w;
#pragma unroll
    for (int o = 16; o > 0; o >>= 1) d += __shfl_down(d, o, 32);
    if (lane == 0) out[node] = d + b[0];
}

// ---------------- host orchestration ----------------

extern "C" void kernel_launch(void* const* d_in, const int* in_sizes, int n_in,
                              void* d_out, int out_size, void* d_ws, size_t ws_size,
                              hipStream_t stream) {
    const float* x        = (const float*)d_in[0];
    const int*   ei       = (const int*)d_in[1];
    const float* sage0_Wl = (const float*)d_in[2];
    const float* sage0_bl = (const float*)d_in[3];
    const float* sage0_Wr = (const float*)d_in[4];
    const float* lin0_W   = (const float*)d_in[5];
    const float* lin0_b   = (const float*)d_in[6];
    const float* bn_gamma = (const float*)d_in[7];
    const float* bn_beta  = (const float*)d_in[8];
    const float* sageh_Wl = (const float*)d_in[9];
    const float* sageh_bl = (const float*)d_in[10];
    const float* sageh_Wr = (const float*)d_in[11];
    const float* linh_W   = (const float*)d_in[12];
    const float* linh_b   = (const float*)d_in[13];
    const float* out_W    = (const float*)d_in[14];
    const float* out_b    = (const float*)d_in[15];

    const int N = in_sizes[0] / D;       // 50000 (multiple of 16)
    const int E = in_sizes[1] / 2;       // 800000
    const int* src = ei;
    const int* dst = ei + E;

    // carve workspace
    char* w = (char*)d_ws;
    auto alloc = [&](size_t bytes) {
        char* p = w;
        w += (bytes + 255) & ~(size_t)255;
        return p;
    };
    float*  agg    = (float*)alloc((size_t)N * D * 4);
    float*  hA     = (float*)alloc((size_t)N * D * 4);
    float*  hB     = (float*)alloc((size_t)N * D * 4);
    float*  deg    = (float*)alloc((size_t)N * 4);
    float*  invdeg = (float*)alloc((size_t)N * 4);
    float*  sums   = (float*)alloc(2 * D * 4);         // sums | sumsq contiguous
    float*  sumsq  = sums + D;
    float*  bc     = (float*)alloc(4 * D * 4);         // combined biases (layer0 + 3 hidden)
    __bf16* packs  = (__bf16*)alloc(8 * (size_t)D * D * 2);

    __bf16* P_Wl0 = packs + 0 * D * D;
    __bf16* P_Wc0 = packs + 1 * D * D;
    // hidden i: Wl at 2+2i, Wc at 3+2i

    const int nRowTiles = N / 16;                       // 3125
    const int gemmBlocks = (nRowTiles + 7) / 8;         // 8 waves/block
    const int edgeWaveBlocks = ((E * 32) + 255) / 256;

    // degree (reused by all 4 aggregations)
    zero_kernel<<<64, 256, 0, stream>>>(deg, (long)N);
    degree_kernel<<<(E + 255) / 256, 256, 0, stream>>>(dst, deg, E);
    invdeg_kernel<<<(N + 255) / 256, 256, 0, stream>>>(deg, invdeg, N);

    // pack weights into WMMA B-fragment layout; fold Wr + parallel-linear W
    pack_weights_kernel<<<4, 256, 0, stream>>>(sage0_Wl, nullptr, P_Wl0);
    pack_weights_kernel<<<4, 256, 0, stream>>>(sage0_Wr, lin0_W, P_Wc0);
    for (int i = 0; i < 3; ++i) {
        pack_weights_kernel<<<4, 256, 0, stream>>>(sageh_Wl + (size_t)i * D * D, nullptr,
                                                   packs + (size_t)(2 + 2 * i) * D * D);
        pack_weights_kernel<<<4, 256, 0, stream>>>(sageh_Wr + (size_t)i * D * D,
                                                   linh_W + (size_t)i * D * D,
                                                   packs + (size_t)(3 + 2 * i) * D * D);
    }
    // combined biases
    vecadd_kernel<<<1, D, 0, stream>>>(sage0_bl, lin0_b, bc, D);
    for (int i = 0; i < 3; ++i)
        vecadd_kernel<<<1, D, 0, stream>>>(sageh_bl + i * D, linh_b + i * D, bc + (1 + i) * D, D);

    // ---- layer 0: SAGE + parallel linear -> BN -> ELU ----
    zero_kernel<<<1024, 256, 0, stream>>>(agg, (long)N * D);
    scatter_kernel<<<edgeWaveBlocks, 256, 0, stream>>>(x, src, dst, agg, E);
    gemm_dual_kernel<<<gemmBlocks, 256, 0, stream>>>(agg, invdeg, x, P_Wl0, P_Wc0, bc, hB,
                                                     nRowTiles, /*elu=*/0);
    zero_kernel<<<1, 256, 0, stream>>>(sums, 2 * D);
    bn_stats_kernel<<<512, D, 0, stream>>>(hB, sums, sumsq, N);
    bn_elu_kernel<<<2048, 256, 0, stream>>>(hB, sums, sumsq, bn_gamma, bn_beta, hA, N);

    // ---- hidden layers ----
    float* cur = hA;
    float* nxt = hB;
    for (int i = 0; i < 3; ++i) {
        zero_kernel<<<1024, 256, 0, stream>>>(agg, (long)N * D);
        scatter_kernel<<<edgeWaveBlocks, 256, 0, stream>>>(cur, src, dst, agg, E);
        gemm_dual_kernel<<<gemmBlocks, 256, 0, stream>>>(
            agg, invdeg, cur, packs + (size_t)(2 + 2 * i) * D * D,
            packs + (size_t)(3 + 2 * i) * D * D, bc + (1 + i) * D, nxt, nRowTiles, /*elu=*/1);
        float* t = cur; cur = nxt; nxt = t;
    }

    // ---- output projection ----
    out_kernel<<<((N * 32) + 255) / 256, 256, 0, stream>>>(cur, out_W, out_b, (float*)d_out, N);
    (void)ws_size; (void)n_in; (void)out_size;
}
